// DeepAR_28303834480654
// MI455X (gfx1250) — matrix operand order repaired and checked by reading, not compiled, not run on hardware
//
#include <hip/hip_runtime.h>
#include <hip/hip_bf16.h>
#include <math.h>

// ---------------- problem constants ----------------
#define BATCH   1024
#define T_ENC   168
#define H_DEC   24
#define ENC_IN  32
#define DEC_IN  16
#define HID     512
#define G4      2048            // 4*HID
#define KT0     544             // ENC_IN + HID        (17 * 32)
#define KT1     1024            // HID + HID           (32 * 32)
#define KTD     1056            // 32 (cov|y|pad) + HID(ctx) + HID(h)  (33 * 32)
#define HALF_GRID 256           // blocks per layer: 2048 waves / 8 per block

typedef __attribute__((ext_vector_type(16))) _Float16 v16h;
typedef __attribute__((ext_vector_type(8)))  _Float16 v8h;
typedef __attribute__((ext_vector_type(8)))  float    v8f;

union V16H { v16h v; v8h h8[2]; _Float16 e[16]; };

__device__ __forceinline__ float sigmoidf_(float x) {
    return 1.0f / (1.0f + __expf(-x));
}

// ---------------- weight / bias preparation ----------------
__global__ void build_w0(const float* __restrict__ Wih, const float* __restrict__ Whh,
                         _Float16* __restrict__ out) {
    int idx = blockIdx.x * 256 + threadIdx.x;
    if (idx >= G4 * KT0) return;
    int r = idx / KT0, c = idx % KT0;
    float v = (c < ENC_IN) ? Wih[r * ENC_IN + c] : Whh[r * HID + (c - ENC_IN)];
    out[idx] = (_Float16)v;
}

__global__ void build_w1(const float* __restrict__ Wih, const float* __restrict__ Whh,
                         _Float16* __restrict__ out) {
    int idx = blockIdx.x * 256 + threadIdx.x;
    if (idx >= G4 * KT1) return;
    int r = idx / KT1, c = idx % KT1;
    float v = (c < HID) ? Wih[r * HID + c] : Whh[r * HID + (c - HID)];
    out[idx] = (_Float16)v;
}

// Wdcat layout: [cov(0..15) | y(16) | zero(17..31) | ctx(32..543) | h(544..1055)]
// W_ihd columns: 0..15 = cov, 16..527 = context, 528 = y_prev
__global__ void build_wd(const float* __restrict__ Wih, const float* __restrict__ Whh,
                         _Float16* __restrict__ out) {
    int idx = blockIdx.x * 256 + threadIdx.x;
    if (idx >= G4 * KTD) return;
    int r = idx / KTD, c = idx % KTD;
    float v;
    if (c < 16)        v = Wih[r * 529 + c];
    else if (c == 16)  v = Wih[r * 529 + 528];
    else if (c < 32)   v = 0.0f;
    else if (c < 544)  v = Wih[r * 529 + (c - 32 + 16)];
    else               v = Whh[r * HID + (c - 544)];
    out[idx] = (_Float16)v;
}

__global__ void build_bias(const float* a0, const float* b0,
                           const float* a1, const float* b1,
                           const float* a2, const float* b2,
                           float* o0, float* o1, float* o2) {
    int i = blockIdx.x * 256 + threadIdx.x;
    if (i < G4)              o0[i]        = a0[i]        + b0[i];
    else if (i < 2 * G4)     o1[i - G4]   = a1[i - G4]   + b1[i - G4];
    else if (i < 3 * G4)     o2[i - 2*G4] = a2[i - 2*G4] + b2[i - 2*G4];
}

// ---------------- A-fragment loader (per K-block of 32) ----------------
template <int MODE>
__device__ __forceinline__ void load_a_frag(
    V16H& a, int kk, int arow, int hi,
    const _Float16* __restrict__ S0, const _Float16* __restrict__ Hprev,
    const float* __restrict__ extf, const float* __restrict__ yprev)
{
    if (MODE == 0) {
        if (kk == 0) {   // encoder input x_t, f32 -> f16 inline; row stride T_ENC*ENC_IN
            const float* p = extf + (size_t)arow * (T_ENC * ENC_IN);
            #pragma unroll
            for (int w = 0; w < 8; ++w) a.e[w]     = (_Float16)p[hi * 8 + w];
            #pragma unroll
            for (int w = 0; w < 8; ++w) a.e[8 + w] = (_Float16)p[16 + hi * 8 + w];
        } else {
            const _Float16* p = Hprev + (size_t)arow * HID + (kk - 32);
            a.h8[0] = *(const v8h*)(p + hi * 8);
            a.h8[1] = *(const v8h*)(p + 16 + hi * 8);
        }
    } else if (MODE == 1) {
        const _Float16* p = (kk < HID) ? (S0 + (size_t)arow * HID + kk)
                                       : (Hprev + (size_t)arow * HID + (kk - HID));
        a.h8[0] = *(const v8h*)(p + hi * 8);
        a.h8[1] = *(const v8h*)(p + 16 + hi * 8);
    } else {
        if (kk == 0) {   // [cov(16) | y(1) | zeros]; dec_inp row stride H_DEC*DEC_IN
            const float* p = extf + (size_t)arow * (H_DEC * DEC_IN);
            #pragma unroll
            for (int w = 0; w < 8; ++w) a.e[w] = (_Float16)p[hi * 8 + w]; // K = hi*8+w in [0,16)
            if (hi == 0) {   // K = 16..23 : y_prev at K==16, rest zero
                float y = yprev ? yprev[(size_t)arow * H_DEC] : 0.0f;
                a.e[8] = (_Float16)y;
                #pragma unroll
                for (int w = 1; w < 8; ++w) a.e[8 + w] = (_Float16)0.0f;
            } else {         // K = 24..31 : zero pad
                #pragma unroll
                for (int w = 0; w < 8; ++w) a.e[8 + w] = (_Float16)0.0f;
            }
        } else if (kk < 544) {   // context
            const _Float16* p = S0 + (size_t)arow * HID + (kk - 32);
            a.h8[0] = *(const v8h*)(p + hi * 8);
            a.h8[1] = *(const v8h*)(p + 16 + hi * 8);
        } else {                  // previous decoder hidden
            const _Float16* p = Hprev + (size_t)arow * HID + (kk - 544);
            a.h8[0] = *(const v8h*)(p + hi * 8);
            a.h8[1] = *(const v8h*)(p + 16 + hi * 8);
        }
    }
}

// ---------------- fused WMMA LSTM tile body ----------------
// One wave owns a 16(batch) x 16(hidden-unit) tile, accumulating all 4 gates
// (i,f,g,o) so the LSTM cell update happens in-register.  2048 waves per layer.
// Wave->tile mapping: the 8 waves of a block share ntile (identical B/weight
// fragments -> L0 hits) and cover 8 consecutive mtiles.
// A/B fragments are double-buffered so next K-block loads issue while the
// current 4 WMMAs execute.
template <int MODE>
__device__ __forceinline__ void lstm_tile_body(
    int blk,                             // layer-local block index [0, HALF_GRID)
    const _Float16* __restrict__ Wcat,   // [G4, Ktot] f16 row-major
    const float*    __restrict__ bias,   // [G4] combined b_ih+b_hh
    const _Float16* __restrict__ S0,     // mode1: h0 current ; mode2: context
    const _Float16* __restrict__ Hprev,  // previous hidden (f16 [B,HID])
    const float*    __restrict__ extf,   // mode0: enc_inp+t*32 ; mode2: dec_inp+t*16
    const float*    __restrict__ yprev,  // mode2: tgt+(t-1) or nullptr (t==0)
    float*          __restrict__ Cst,    // cell state f32 [B,HID], in-place
    _Float16*       __restrict__ Hnew)   // new hidden f16 [B,HID]
{
    const int KT   = (MODE == 0) ? KT0 : (MODE == 1) ? KT1 : KTD;
    const int wid  = (blk * 256 + (int)threadIdx.x) >> 5;
    const int lane = threadIdx.x & 31;
    const int ln   = lane & 15;
    const int hi   = lane >> 4;
    const int mtile = wid & 63;          // 64 tiles of 16 batch rows (varies in block)
    const int ntile = wid >> 6;          // 32 tiles of 16 hidden units (fixed in block)
    const int col   = ntile * 16 + ln;   // hidden unit index (B-matrix N col)
    const int arow  = mtile * 16 + ln;   // A-matrix M row (batch index)

    v8f accI, accF, accG, accO;
    {
        float bi = bias[col], bf = bias[col + HID];
        float bg = bias[col + 2 * HID], bo = bias[col + 3 * HID];
        #pragma unroll
        for (int r = 0; r < 8; ++r) { accI[r] = bi; accF[r] = bf; accG[r] = bg; accO[r] = bo; }
    }

    // weight row bases for the 4 gates (B[k][n] = Wcat[gate*HID + col][k])
    const _Float16* wI = Wcat + (size_t)(0 * HID + col) * KT + hi * 8;
    const _Float16* wF = Wcat + (size_t)(1 * HID + col) * KT + hi * 8;
    const _Float16* wG = Wcat + (size_t)(2 * HID + col) * KT + hi * 8;
    const _Float16* wO = Wcat + (size_t)(3 * HID + col) * KT + hi * 8;

    // prologue: fragments for kk = 0
    V16H a;
    load_a_frag<MODE>(a, 0, arow, hi, S0, Hprev, extf, yprev);
    v16h bI = __builtin_shufflevector(*(const v8h*)(wI), *(const v8h*)(wI + 16), 0,1,2,3,4,5,6,7,8,9,10,11,12,13,14,15);
    v16h bF = __builtin_shufflevector(*(const v8h*)(wF), *(const v8h*)(wF + 16), 0,1,2,3,4,5,6,7,8,9,10,11,12,13,14,15);
    v16h bG = __builtin_shufflevector(*(const v8h*)(wG), *(const v8h*)(wG + 16), 0,1,2,3,4,5,6,7,8,9,10,11,12,13,14,15);
    v16h bO = __builtin_shufflevector(*(const v8h*)(wO), *(const v8h*)(wO + 16), 0,1,2,3,4,5,6,7,8,9,10,11,12,13,14,15);

    for (int kk = 0; kk < KT; kk += 32) {
        const int kn = kk + 32;
        V16H a2;
        v16h bI2, bF2, bG2, bO2;
        if (kn < KT) {   // prefetch next K-block while current WMMAs run
            load_a_frag<MODE>(a2, kn, arow, hi, S0, Hprev, extf, yprev);
            bI2 = __builtin_shufflevector(*(const v8h*)(wI + kn), *(const v8h*)(wI + kn + 16), 0,1,2,3,4,5,6,7,8,9,10,11,12,13,14,15);
            bF2 = __builtin_shufflevector(*(const v8h*)(wF + kn), *(const v8h*)(wF + kn + 16), 0,1,2,3,4,5,6,7,8,9,10,11,12,13,14,15);
            bG2 = __builtin_shufflevector(*(const v8h*)(wG + kn), *(const v8h*)(wG + kn + 16), 0,1,2,3,4,5,6,7,8,9,10,11,12,13,14,15);
            bO2 = __builtin_shufflevector(*(const v8h*)(wO + kn), *(const v8h*)(wO + kn + 16), 0,1,2,3,4,5,6,7,8,9,10,11,12,13,14,15);
        }
        accI = __builtin_amdgcn_wmma_f32_16x16x32_f16(false, a.v, false, bI, (short)0, accI, false, false);
        accF = __builtin_amdgcn_wmma_f32_16x16x32_f16(false, a.v, false, bF, (short)0, accF, false, false);
        accG = __builtin_amdgcn_wmma_f32_16x16x32_f16(false, a.v, false, bG, (short)0, accG, false, false);
        accO = __builtin_amdgcn_wmma_f32_16x16x32_f16(false, a.v, false, bO, (short)0, accO, false, false);
        if (kn < KT) {
            a = a2; bI = bI2; bF = bF2; bG = bG2; bO = bO2;
        }
    }

    // LSTM cell update, in-register.  C/D layout: VGPR r -> M = r + hi*8.
    const int rbase = mtile * 16 + hi * 8;
    #pragma unroll
    for (int r = 0; r < 8; ++r) {
        size_t idx = (size_t)(rbase + r) * HID + col;
        float iv = sigmoidf_(accI[r]);
        float fv = sigmoidf_(accF[r]);
        float gv = tanhf(accG[r]);
        float ov = sigmoidf_(accO[r]);
        float c  = fv * Cst[idx] + iv * gv;
        Cst[idx] = c;
        Hnew[idx] = (_Float16)(ov * tanhf(c));
    }
}

// ---------------- skew-pipelined dual encoder step ----------------
// Launch L runs layer0 step L (blocks [0,256)) and layer1 step L-1 (blocks
// [256,512)) concurrently: layer1 step L-1 depends only on state produced by
// launch L-1, so the two halves are independent.  Halves the serial launch
// chain (169 launches instead of 336) and doubles per-launch parallelism.
__global__ __launch_bounds__(256) void enc_dual(
    const _Float16* __restrict__ W0, const float* __restrict__ Bi0,
    const _Float16* __restrict__ h0r, _Float16* __restrict__ h0w,
    const float* __restrict__ enc_t, float* __restrict__ C0, int do0,
    const _Float16* __restrict__ W1, const float* __restrict__ Bi1,
    const _Float16* __restrict__ s0,  const _Float16* __restrict__ h1r,
    _Float16* __restrict__ h1w, float* __restrict__ C1, int do1)
{
    if (blockIdx.x < HALF_GRID) {
        if (do0) lstm_tile_body<0>(blockIdx.x, W0, Bi0, nullptr, h0r, enc_t, nullptr, C0, h0w);
    } else {
        if (do1) lstm_tile_body<1>(blockIdx.x - HALF_GRID, W1, Bi1, s0, h1r, nullptr, nullptr, C1, h1w);
    }
}

// ---------------- decoder cell step ----------------
__global__ __launch_bounds__(256) void dec_step(
    const _Float16* __restrict__ Wd, const float* __restrict__ Bid,
    const _Float16* __restrict__ ctx, const _Float16* __restrict__ hdr,
    const float* __restrict__ dec_t, const float* __restrict__ yprev,
    float* __restrict__ Cd, _Float16* __restrict__ hdw)
{
    lstm_tile_body<2>(blockIdx.x, Wd, Bid, ctx, hdr, dec_t, yprev, Cd, hdw);
}

// ---------------- output heads: mu / softplus(sigma) ----------------
__global__ __launch_bounds__(256) void head_kernel(
    const _Float16* __restrict__ Hd, const float* __restrict__ Wmu, const float* __restrict__ bmu,
    const float* __restrict__ Wsig, const float* __restrict__ bsig,
    float* __restrict__ out, int t)
{
    int row  = (blockIdx.x * blockDim.x + threadIdx.x) >> 5;
    int lane = threadIdx.x & 31;
    if (row >= BATCH) return;
    const _Float16* h = Hd + (size_t)row * HID;
    float smu = 0.0f, ssg = 0.0f;
    for (int k = lane; k < HID; k += 32) {
        float hv = (float)h[k];
        smu += hv * Wmu[k];
        ssg += hv * Wsig[k];
    }
    #pragma unroll
    for (int off = 16; off > 0; off >>= 1) {
        smu += __shfl_down(smu, off, 32);
        ssg += __shfl_down(ssg, off, 32);
    }
    if (lane == 0) {
        float x  = ssg + bsig[0];
        float sp = (x > 20.0f) ? x : log1pf(__expf(x));
        out[(size_t)row * H_DEC + t]                          = smu + bmu[0];
        out[(size_t)BATCH * H_DEC + (size_t)row * H_DEC + t]  = sp + 1e-6f;
    }
}

// ---------------- workspace layout (bytes) ----------------
#define OFF_W0     ((size_t)0)
#define SZ_W0      ((size_t)G4 * KT0 * 2)
#define OFF_W1     (OFF_W0 + SZ_W0)
#define SZ_W1      ((size_t)G4 * KT1 * 2)
#define OFF_WD     (OFF_W1 + SZ_W1)
#define SZ_WD      ((size_t)G4 * KTD * 2)
#define OFF_B0     (OFF_WD + SZ_WD)
#define OFF_B1     (OFF_B0 + (size_t)G4 * 4)
#define OFF_BD     (OFF_B1 + (size_t)G4 * 4)
#define SZ_H       ((size_t)BATCH * HID * 2)   // f16 hidden
#define SZ_C       ((size_t)BATCH * HID * 4)   // f32 cell
#define OFF_H00    (OFF_BD + (size_t)G4 * 4)   // ---- zero region start ----
#define OFF_H10    (OFF_H00 + SZ_H)
#define OFF_HD0    (OFF_H10 + SZ_H)
#define OFF_C0     (OFF_HD0 + SZ_H)
#define OFF_C1     (OFF_C0 + SZ_C)
#define OFF_CD     (OFF_C1 + SZ_C)
#define OFF_ZEND   (OFF_CD + SZ_C)             // ---- zero region end ----
#define OFF_H01    (OFF_ZEND)
#define OFF_H11    (OFF_H01 + SZ_H)
#define OFF_HD1    (OFF_H11 + SZ_H)

extern "C" void kernel_launch(void* const* d_in, const int* in_sizes, int n_in,
                              void* d_out, int out_size, void* d_ws, size_t ws_size,
                              hipStream_t stream) {
    const float* enc   = (const float*)d_in[0];
    const float* dec   = (const float*)d_in[1];
    const float* tgt   = (const float*)d_in[2];
    const float* Wih0  = (const float*)d_in[3];
    const float* Whh0  = (const float*)d_in[4];
    const float* bih0  = (const float*)d_in[5];
    const float* bhh0  = (const float*)d_in[6];
    const float* Wih1  = (const float*)d_in[7];
    const float* Whh1  = (const float*)d_in[8];
    const float* bih1  = (const float*)d_in[9];
    const float* bhh1  = (const float*)d_in[10];
    const float* Wihd  = (const float*)d_in[11];
    const float* Whhd  = (const float*)d_in[12];
    const float* bihd  = (const float*)d_in[13];
    const float* bhhd  = (const float*)d_in[14];
    const float* Wmu   = (const float*)d_in[15];
    const float* bmu   = (const float*)d_in[16];
    const float* Wsig  = (const float*)d_in[17];
    const float* bsig  = (const float*)d_in[18];

    char* ws = (char*)d_ws;
    _Float16* W0c = (_Float16*)(ws + OFF_W0);
    _Float16* W1c = (_Float16*)(ws + OFF_W1);
    _Float16* Wdc = (_Float16*)(ws + OFF_WD);
    float*    B0  = (float*)(ws + OFF_B0);
    float*    B1  = (float*)(ws + OFF_B1);
    float*    BD  = (float*)(ws + OFF_BD);
    _Float16* h0[2] = { (_Float16*)(ws + OFF_H00), (_Float16*)(ws + OFF_H01) };
    _Float16* h1[2] = { (_Float16*)(ws + OFF_H10), (_Float16*)(ws + OFF_H11) };
    _Float16* hd[2] = { (_Float16*)(ws + OFF_HD0), (_Float16*)(ws + OFF_HD1) };
    float* C0 = (float*)(ws + OFF_C0);
    float* C1 = (float*)(ws + OFF_C1);
    float* CD = (float*)(ws + OFF_CD);

    // zero initial hidden + cell state (contiguous region), every call
    hipMemsetAsync(ws + OFF_H00, 0, OFF_ZEND - OFF_H00, stream);

    // weight / bias prep
    build_w0<<<(G4 * KT0 + 255) / 256, 256, 0, stream>>>(Wih0, Whh0, W0c);
    build_w1<<<(G4 * KT1 + 255) / 256, 256, 0, stream>>>(Wih1, Whh1, W1c);
    build_wd<<<(G4 * KTD + 255) / 256, 256, 0, stream>>>(Wihd, Whhd, Wdc);
    build_bias<<<(3 * G4 + 255) / 256, 256, 0, stream>>>(bih0, bhh0, bih1, bhh1, bihd, bhhd, B0, B1, BD);

    // ---- skew-pipelined encoder: launch L = { layer0 step L , layer1 step L-1 } ----
    // layer0 step t: reads h0[t&1], writes h0[(t+1)&1]          (h0[0] zeroed)
    // layer1 step u: reads S0 = h0[(u+1)&1], h1[u&1]; writes h1[(u+1)&1]
    // Within launch L both halves read h0[L&1]; only layer0 writes (h0[(L+1)&1]).
    for (int L = 0; L <= T_ENC; ++L) {
        const int do0 = (L < T_ENC) ? 1 : 0;
        const int do1 = (L >= 1) ? 1 : 0;
        const int u   = (L >= 1) ? (L - 1) : 0;   // layer1 time index
        enc_dual<<<2 * HALF_GRID, 256, 0, stream>>>(
            W0c, B0, h0[L & 1], h0[(L + 1) & 1],
            enc + (size_t)(do0 ? L : 0) * ENC_IN, C0, do0,
            W1c, B1, h0[L & 1], h1[u & 1], h1[(u + 1) & 1], C1, do1);
    }
    const _Float16* ctx = h1[0];   // layer1 step 167 wrote h1[(167+1)&1] = h1[0]

    // ---- decoder with teacher forcing ----
    int rd = 0;
    for (int t = 0; t < H_DEC; ++t) {
        const float* yp = (t > 0) ? (tgt + (t - 1)) : nullptr;
        dec_step<<<HALF_GRID, 256, 0, stream>>>(
            Wdc, BD, ctx, hd[rd], dec + (size_t)t * DEC_IN, yp, CD, hd[rd ^ 1]);
        head_kernel<<<(BATCH * 32 + 255) / 256, 256, 0, stream>>>(
            hd[rd ^ 1], Wmu, bmu, Wsig, bsig, (float*)d_out, t);
        rd ^= 1;
    }
}
